// End2EndModel_60284160966886
// MI455X (gfx1250) — matrix operand order repaired and checked by reading, not compiled, not run on hardware
//
#include <hip/hip_runtime.h>
#include <math.h>

#define DEVFN __device__ __forceinline__

typedef __attribute__((ext_vector_type(16))) _Float16 v16h;
typedef __attribute__((ext_vector_type(8)))  _Float16 v8h;
typedef __attribute__((ext_vector_type(8)))  float    v8f;

constexpr int B_ = 128, P_ = 16, NP_ = 256, EP_ = 1280;
constexpr int EPG_ = 2048, ODIM_ = 256;
constexpr float EPSF = 1e-5f;

// ---------------- WMMA helpers (gfx1250, wave32) ----------------

DEVFN v8f vzero8() { v8f z = {}; return z; }

DEVFN v8f wmma16(v16h a, v16h b, v8f c) {
  // emits v_wmma_f32_16x16x32_f16
  return __builtin_amdgcn_wmma_f32_16x16x32_f16(false, a, false, b, (short)0, c,
                                                false, false);
}

// A fragment from row-major [M][lda] f16: lanes 0-15: M=lane, K 0..15;
// lanes 16-31: same M, K 16..31. Two aligned 16B LDS loads.
DEVFN v16h frag_a(const _Float16* A, int lda, int row0, int k0, int lane) {
  const int m  = row0 + (lane & 15);
  const int kg = (lane >> 4) << 4;  // 0 or 16
  const _Float16* r = A + m * lda + k0 + kg;
  v8h lo = *(const v8h*)r;
  v8h hi = *(const v8h*)(r + 8);
  v16h f;
#pragma unroll
  for (int i = 0; i < 8; ++i) { f[i] = lo[i]; f[8 + i] = hi[i]; }
  return f;
}

// B fragment from a pre-swizzled buffer: tile = 32 lanes x 16 contiguous halves
// (lane L holds N=n0+(L&15), K = kt*32 + (L>=16?16:0) + 0..15). Two 16B loads.
DEVFN v16h frag_b_sw(const _Float16* Bsw, int tileIdx, int lane) {
  const _Float16* r = Bsw + (((tileIdx << 5) + lane) << 4);
  v8h lo = *(const v8h*)r;
  v8h hi = *(const v8h*)(r + 8);
  v16h f;
#pragma unroll
  for (int i = 0; i < 8; ++i) { f[i] = lo[i]; f[8 + i] = hi[i]; }
  return f;
}

// C/D: lane L -> N = L&15, M = (L>=16 ? 8 : 0) + vgpr_index
DEVFN void store_c_f32(float* C, int ldc, int row0, int col0, int lane, v8f acc) {
  const int n  = col0 + (lane & 15);
  const int m0 = row0 + ((lane >> 4) << 3);
#pragma unroll
  for (int r = 0; r < 8; ++r) C[(m0 + r) * ldc + n] = acc[r];
}

DEVFN void store_c_f16(_Float16* C, int ldc, int row0, int col0, int lane, v8f acc) {
  const int n  = col0 + (lane & 15);
  const int m0 = row0 + ((lane >> 4) << 3);
#pragma unroll
  for (int r = 0; r < 8; ++r) C[(m0 + r) * ldc + n] = (_Float16)acc[r];
}

// order-preserving float<->u32 key for atomic max on LDS
DEVFN unsigned f2key(float f) {
  unsigned u = __float_as_uint(f);
  return (u & 0x80000000u) ? ~u : (u | 0x80000000u);
}
DEVFN float key2f(unsigned k) {
  unsigned u = (k & 0x80000000u) ? (k & 0x7fffffffu) : ~k;
  return __uint_as_float(u);
}
DEVFN float lrelu(float x) { return x > 0.f ? x : 0.2f * x; }

// async global->LDS 4B per lane (ASYNCcnt); lds_off is byte offset in LDS
DEVFN void async_g2l_b32(unsigned lds_off, const void* gaddr) {
  asm volatile("global_load_async_to_lds_b32 %0, %1, off"
               :: "v"(lds_off), "v"(gaddr) : "memory");
}
DEVFN void wait_async0() { asm volatile("s_wait_asynccnt 0" ::: "memory"); }

// ---------------- Kernel 1: per-(patient,plane) graph encoder ----------------
// LDS layout (bytes)
constexpr int OFF_FEAT = 0;                     // 256 f32
constexpr int OFF_W1   = OFF_FEAT + 1024;       // 128 f32
constexpr int OFF_RES1 = OFF_W1   + 512;
constexpr int OFF_B1   = OFF_RES1 + 512;
constexpr int OFF_G1   = OFF_B1   + 512;
constexpr int OFF_BE1  = OFF_G1   + 512;
constexpr int OFF_ADOT = OFF_BE1  + 512;        // 4 f32
constexpr int OFF_M1   = OFF_ADOT + 16;         // 512 u32
constexpr int OFF_S1   = OFF_M1   + 2048;       // 512 f32
constexpr int OFF_ACC1 = OFF_S1   + 2048;       // 512 f32
constexpr int OFF_ESRC = OFF_ACC1 + 2048;       // 1280 i32
constexpr int OFF_EDST = OFF_ESRC + 5120;       // 1280 i32
constexpr int OFF_H1H  = OFF_EDST + 5120;       // 256x128 f16 (aliased by Dh)
constexpr int OFF_W2R  = OFF_H1H  + 65536;      // [W2|res2] swizzled, 8192 f16
constexpr int OFF_C    = OFF_W2R  + 16384;      // 256x64 f32
constexpr int OFF_AL2  = OFF_C    + 65536;      // 5 x 32 f32
constexpr int OFF_EL2  = OFF_AL2  + 640;        // 256 f32
constexpr int OFF_ER2  = OFF_EL2  + 1024;
constexpr int OFF_M2   = OFF_ER2  + 1024;       // 256 u32
constexpr int OFF_S2   = OFF_M2   + 1024;       // 256 f32
constexpr int OFF_RST2 = OFF_S2   + 1024;       // 256x32 f32
constexpr int OFF_H2H  = OFF_RST2 + 32768;      // 256x32 f16
constexpr int OFF_DW1H = OFF_H2H  + 16384;      // dW1 swizzled, 4096 f16
constexpr int OFF_DW2  = OFF_DW1H + 8192;       // 128 f32
constexpr int OFF_DB1  = OFF_DW2  + 512;
constexpr int OFF_DG   = OFF_DB1  + 512;
constexpr int OFF_DBE  = OFF_DG   + 512;
constexpr int OFF_RED  = OFF_DBE  + 512;        // 256 f32
constexpr int K1_SMEM  = OFF_RED  + 1024;       // 232592 B (< 320KB/WGP)

__global__ __launch_bounds__(256) void plane_encoder_kernel(
    const float* __restrict__ plane_feat, const int* __restrict__ plane_src,
    const int* __restrict__ plane_dst,
    const float* __restrict__ W1, const float* __restrict__ al1,
    const float* __restrict__ ar1, const float* __restrict__ res1,
    const float* __restrict__ b1, const float* __restrict__ g1,
    const float* __restrict__ be1,
    const float* __restrict__ W2, const float* __restrict__ al2,
    const float* __restrict__ ar2, const float* __restrict__ res2,
    const float* __restrict__ b2, const float* __restrict__ g2,
    const float* __restrict__ be2,
    const float* __restrict__ dW1, const float* __restrict__ db1,
    const float* __restrict__ dg, const float* __restrict__ dbe,
    const float* __restrict__ dW2, const float* __restrict__ db2,
    float* __restrict__ ws_rep, float* __restrict__ ws_rloss) {
  extern __shared__ char smem[];
  const int tid  = threadIdx.x;
  const int lane = tid & 31;
  const int wave = tid >> 5;
  const int g    = blockIdx.x;   // graph id = b*P + p
  const int p    = g % P_;

  float*     feat  = (float*)(smem + OFF_FEAT);
  float*     W1v   = (float*)(smem + OFF_W1);
  float*     res1v = (float*)(smem + OFF_RES1);
  float*     b1v   = (float*)(smem + OFF_B1);
  float*     g1v   = (float*)(smem + OFF_G1);
  float*     be1v  = (float*)(smem + OFF_BE1);
  float*     adot  = (float*)(smem + OFF_ADOT);
  unsigned*  m1    = (unsigned*)(smem + OFF_M1);
  float*     s1    = (float*)(smem + OFF_S1);
  float*     acc1  = (float*)(smem + OFF_ACC1);
  int*       esrc  = (int*)(smem + OFF_ESRC);
  int*       edst  = (int*)(smem + OFF_EDST);
  _Float16*  h1h   = (_Float16*)(smem + OFF_H1H);   // [256][128]
  _Float16*  Dh    = (_Float16*)(smem + OFF_H1H);   // alias: decoder act
  _Float16*  w2r   = (_Float16*)(smem + OFF_W2R);   // swizzled [W2|res2]
  float*     Cm    = (float*)(smem + OFF_C);        // [256][64] = [z2 | res]
  float*     al2v  = (float*)(smem + OFF_AL2);
  float*     ar2v  = al2v + 32;
  float*     b2v   = al2v + 64;
  float*     g2v   = al2v + 96;
  float*     be2v  = al2v + 128;
  float*     el2   = (float*)(smem + OFF_EL2);
  float*     er2   = (float*)(smem + OFF_ER2);
  unsigned*  m2    = (unsigned*)(smem + OFF_M2);
  float*     s2    = (float*)(smem + OFF_S2);
  float*     rst2  = (float*)(smem + OFF_RST2);     // [256][32]
  _Float16*  h2h   = (_Float16*)(smem + OFF_H2H);   // [256][32]
  _Float16*  dW1h  = (_Float16*)(smem + OFF_DW1H);  // swizzled dW1
  float*     dW2v  = (float*)(smem + OFF_DW2);
  float*     db1v  = (float*)(smem + OFF_DB1);
  float*     dgv   = (float*)(smem + OFF_DG);
  float*     dbev  = (float*)(smem + OFF_DBE);
  float*     red   = (float*)(smem + OFF_RED);

  // prefetch parameter blocks for this plane (global_prefetch_b8)
  __builtin_prefetch(W2 + p * 4096, 0, 3);
  __builtin_prefetch(res2 + p * 4096, 0, 3);
  __builtin_prefetch(dW1 + p * 4096, 0, 3);

  // async global->LDS edge lists (ASYNCcnt path)
  for (int k = tid; k < EP_; k += 256) {
    async_g2l_b32((unsigned)(OFF_ESRC + k * 4), plane_src + (size_t)g * EP_ + k);
    async_g2l_b32((unsigned)(OFF_EDST + k * 4), plane_dst + (size_t)g * EP_ + k);
  }

  // ---- load feat + GAT1 params ----
  feat[tid] = plane_feat[(size_t)g * NP_ + tid];
  if (tid < 128) {
    W1v[tid]   = W1[p * 128 + tid];
    res1v[tid] = res1[p * 128 + tid];
    b1v[tid]   = b1[p * 128 + tid];
    g1v[tid]   = g1[p * 128 + tid];
    be1v[tid]  = be1[p * 128 + tid];
  }
  __syncthreads();
  if (tid < 2) {  // per-head attention dot with W1 row (z factorizes)
    float sa = 0.f, sr = 0.f;
    for (int d = 0; d < 64; ++d) {
      float w = W1v[tid * 64 + d];
      sa += w * al1[p * 128 + tid * 64 + d];
      sr += w * ar1[p * 128 + tid * 64 + d];
    }
    adot[tid] = sa;
    adot[2 + tid] = sr;
  }
  for (int i = tid; i < 512; i += 256) { m1[i] = 0u; s1[i] = 0.f; acc1[i] = 0.f; }
  wait_async0();
  __syncthreads();

  // ---- GAT1 edge softmax (factorized: only 2 head scalars per node) ----
  for (int k = tid; k < EP_; k += 256) {
    int s = esrc[k], d = edst[k];
    float fs = feat[s], fd = feat[d];
#pragma unroll
    for (int h = 0; h < 2; ++h) {
      float e = lrelu(fs * adot[h] + fd * adot[2 + h]);
      atomicMax(&m1[d * 2 + h], f2key(e));
    }
  }
  __syncthreads();
  for (int k = tid; k < EP_; k += 256) {
    int s = esrc[k], d = edst[k];
    float fs = feat[s], fd = feat[d];
#pragma unroll
    for (int h = 0; h < 2; ++h) {
      float e = lrelu(fs * adot[h] + fd * adot[2 + h]);
      atomicAdd(&s1[d * 2 + h], __expf(e - key2f(m1[d * 2 + h])));
    }
  }
  __syncthreads();
  for (int k = tid; k < EP_; k += 256) {
    int s = esrc[k], d = edst[k];
    float fs = feat[s], fd = feat[d];
#pragma unroll
    for (int h = 0; h < 2; ++h) {
      float e  = lrelu(fs * adot[h] + fd * adot[2 + h]);
      float a  = __expf(e - key2f(m1[d * 2 + h])) / (s1[d * 2 + h] + 1e-9f);
      atomicAdd(&acc1[d * 2 + h], a * fs);
    }
  }
  __syncthreads();

  // ---- BN1 + ReLU -> h1 (f16). h1[n][c] = acc[n][h]*W1[c] + feat*res1 + b1 ----
  if (tid < 128) {
    const int c = tid, h = c >> 6;
    const float w = W1v[c], r = res1v[c], bb = b1v[c];
    float sum = 0.f, sq = 0.f;
    for (int n = 0; n < 256; ++n) {
      float x = acc1[n * 2 + h] * w + feat[n] * r + bb;
      sum += x; sq += x * x;
    }
    float mu  = sum * (1.f / 256.f);
    float var = sq * (1.f / 256.f) - mu * mu;
    float inv = rsqrtf(var + EPSF);
    float gg = g1v[c], bw = be1v[c];
    for (int n = 0; n < 256; ++n) {
      float x = acc1[n * 2 + h] * w + feat[n] * r + bb;
      float y = gg * (x - mu) * inv + bw;
      h1h[n * 128 + c] = (_Float16)(y > 0.f ? y : 0.f);
    }
  }
  // stage fused [W2 | res2] into swizzled fragment layout (K=128,N=64 -> NT=4)
  for (int idx = tid; idx < 8192; idx += 256) {
    int t = idx >> 9, ln = (idx >> 4) & 31, i = idx & 15;
    int kt = t >> 2, nt = t & 3;
    int k = kt * 32 + ((ln >> 4) << 4) + i;
    int n = nt * 16 + (ln & 15);
    float v = (n < 32) ? W2[p * 4096 + k * 32 + n] : res2[p * 4096 + k * 32 + (n - 32)];
    w2r[idx] = (_Float16)v;
  }
  __syncthreads();

  // ---- GEMM1 (WMMA): [256,128] x [128,64] -> C [z2 | res] ----
  for (int i = 0; i < 8; ++i) {
    int t = wave * 8 + i;
    int mt = t >> 2, nt = t & 3;
    v8f acc = vzero8();
#pragma unroll
    for (int kt = 0; kt < 4; ++kt) {
      v16h a  = frag_a(h1h, 128, mt * 16, kt * 32, lane);
      v16h bf = frag_b_sw(w2r, kt * 4 + nt, lane);
      acc = wmma16(a, bf, acc);
    }
    store_c_f32(Cm, 64, mt * 16, nt * 16, lane, acc);
  }
  __syncthreads();

  // ---- GAT2 (1 head, 32 dims) ----
  if (tid < 32) {
    al2v[tid] = al2[p * 32 + tid];
    ar2v[tid] = ar2[p * 32 + tid];
    b2v[tid]  = b2[p * 32 + tid];
    g2v[tid]  = g2[p * 32 + tid];
    be2v[tid] = be2[p * 32 + tid];
  }
  __syncthreads();
  {
    const int n = tid;
    float sl = 0.f, sr = 0.f;
    for (int j = 0; j < 32; ++j) {
      float z = Cm[n * 64 + j];
      sl += z * al2v[j]; sr += z * ar2v[j];
    }
    el2[n] = sl; er2[n] = sr; m2[n] = 0u; s2[n] = 0.f;
  }
  for (int idx = tid; idx < 256 * 32; idx += 256) rst2[idx] = 0.f;
  __syncthreads();
  for (int k = tid; k < EP_; k += 256) {
    int s = esrc[k], d = edst[k];
    atomicMax(&m2[d], f2key(lrelu(el2[s] + er2[d])));
  }
  __syncthreads();
  for (int k = tid; k < EP_; k += 256) {
    int s = esrc[k], d = edst[k];
    float e = lrelu(el2[s] + er2[d]);
    atomicAdd(&s2[d], __expf(e - key2f(m2[d])));
  }
  __syncthreads();
  // wave-per-edge, lane-per-channel: 32 lanes stripe 32 LDS banks
  for (int k = wave; k < EP_; k += 8) {
    int s = esrc[k], d = edst[k];
    float e = lrelu(el2[s] + er2[d]);
    float a = __expf(e - key2f(m2[d])) / (s2[d] + 1e-9f);
    atomicAdd(&rst2[d * 32 + lane], a * Cm[s * 64 + lane]);
  }
  __syncthreads();

  // ---- BN2 + ReLU -> h2 (f16); rep = mean over nodes ----
  if (tid < 32) {
    const int c = tid;
    const float bb = b2v[c];
    float sum = 0.f, sq = 0.f;
    for (int n = 0; n < 256; ++n) {
      float x = rst2[n * 32 + c] + Cm[n * 64 + 32 + c] + bb;
      sum += x; sq += x * x;
    }
    float mu  = sum * (1.f / 256.f);
    float var = sq * (1.f / 256.f) - mu * mu;
    float inv = rsqrtf(var + EPSF);
    float gg = g2v[c], bw = be2v[c], rsum = 0.f;
    for (int n = 0; n < 256; ++n) {
      float x = rst2[n * 32 + c] + Cm[n * 64 + 32 + c] + bb;
      float y = gg * (x - mu) * inv + bw;
      y = y > 0.f ? y : 0.f;
      h2h[n * 32 + c] = (_Float16)y;
      rsum += y;
    }
    ws_rep[(size_t)g * 32 + c] = rsum * (1.f / 256.f);
  }
  // stage dW1 swizzled (K=32,N=128 -> KT=1,NT=8) + decoder vectors
  for (int idx = tid; idx < 4096; idx += 256) {
    int t = idx >> 9, ln = (idx >> 4) & 31, i = idx & 15;
    int k = ((ln >> 4) << 4) + i;
    int n = t * 16 + (ln & 15);
    dW1h[idx] = (_Float16)dW1[p * 4096 + k * 128 + n];
  }
  if (tid < 128) {
    dW2v[tid] = dW2[p * 128 + tid];
    db1v[tid] = db1[p * 128 + tid];
    dgv[tid]  = dg[p * 128 + tid];
    dbev[tid] = dbe[p * 128 + tid];
  }
  __syncthreads();

  // ---- GEMM2 (WMMA): [256,32] x [32,128] -> Dh (aliases dead h1h) ----
  for (int i = 0; i < 16; ++i) {
    int t = wave * 16 + i;
    int mt = t >> 3, nt = t & 7;
    v16h a  = frag_a(h2h, 32, mt * 16, 0, lane);
    v16h bf = frag_b_sw(dW1h, nt, lane);
    v8f acc = vzero8();
    acc = wmma16(a, bf, acc);
    store_c_f16(Dh, 128, mt * 16, nt * 16, lane, acc);
  }
  __syncthreads();

  // ---- decoder BN + ReLU in place ----
  if (tid < 128) {
    const int c = tid;
    const float bb = db1v[c];
    float sum = 0.f, sq = 0.f;
    for (int n = 0; n < 256; ++n) {
      float x = (float)Dh[n * 128 + c] + bb;
      sum += x; sq += x * x;
    }
    float mu  = sum * (1.f / 256.f);
    float var = sq * (1.f / 256.f) - mu * mu;
    float inv = rsqrtf(var + EPSF);
    float gg = dgv[c], bw = dbev[c];
    for (int n = 0; n < 256; ++n) {
      float x = (float)Dh[n * 128 + c] + bb;
      float y = gg * (x - mu) * inv + bw;
      Dh[n * 128 + c] = (_Float16)(y > 0.f ? y : 0.f);
    }
  }
  __syncthreads();

  // ---- recon + mse ----
  {
    const int n = tid;
    float acc = db2[p];
    for (int c = 0; c < 128; ++c) acc += (float)Dh[n * 128 + c] * dW2v[c];
    float d = acc - feat[n];
    red[tid] = d * d;
  }
  __syncthreads();
  for (int off = 128; off > 0; off >>= 1) {
    if (tid < off) red[tid] += red[tid + off];
    __syncthreads();
  }
  if (tid == 0) ws_rloss[g] = red[0] * (1.f / 256.f);
}

// ---------------- Kernel 2: patient-level head (single workgroup) ----------------
constexpr int K2_H    = 0;                    // 128x128 f32
constexpr int K2_HSUM = K2_H    + 65536;
constexpr int K2_TMP  = K2_HSUM + 65536;      // gemm out / agg / z (aliased)
constexpr int K2_BUFA = K2_TMP  + 65536;      // 128x128 f16
constexpr int K2_BUFB = K2_BUFA + 32768;      // swizzled B, 128x128 f16
constexpr int K2_MASK = K2_BUFB + 32768;      // 128 f32
constexpr int K2_DEGO = K2_MASK + 512;
constexpr int K2_DEGI = K2_DEGO + 512;
constexpr int K2_PSRC = K2_DEGI + 512;        // 2048 i32
constexpr int K2_PDST = K2_PSRC + 8192;
constexpr int K2_RED  = K2_PDST + 8192;       // 256 f32
constexpr int K2_SMEM = K2_RED  + 1024;       // 281088 B

__global__ __launch_bounds__(256) void patient_head_kernel(
    const float* __restrict__ orig, const int* __restrict__ patient_src,
    const int* __restrict__ patient_dst, const int* __restrict__ mask,
    const float* __restrict__ ft_W, const float* __restrict__ ft_b,
    const float* __restrict__ ft_g, const float* __restrict__ ft_be,
    const float* __restrict__ gc_W, const float* __restrict__ gc_b,
    const float* __restrict__ gbn_g, const float* __restrict__ gbn_be,
    const float* __restrict__ c1_W, const float* __restrict__ c1_b,
    const float* __restrict__ ln_g, const float* __restrict__ ln_be,
    const float* __restrict__ c2_W, const float* __restrict__ c2_b,
    const float* __restrict__ ws_rep, const float* __restrict__ ws_rloss,
    float* __restrict__ out) {
  extern __shared__ char smem[];
  const int tid  = threadIdx.x;
  const int lane = tid & 31;
  const int wave = tid >> 5;

  float*    h     = (float*)(smem + K2_H);
  float*    hsum  = (float*)(smem + K2_HSUM);
  float*    tmp   = (float*)(smem + K2_TMP);
  float*    z     = tmp;  // alias (128x64 after classifier gemm)
  _Float16* bufA  = (_Float16*)(smem + K2_BUFA);
  _Float16* bufB  = (_Float16*)(smem + K2_BUFB);
  float*    maskf = (float*)(smem + K2_MASK);
  float*    dego  = (float*)(smem + K2_DEGO);
  float*    degi  = (float*)(smem + K2_DEGI);
  int*      psrc  = (int*)(smem + K2_PSRC);
  int*      pdst  = (int*)(smem + K2_PDST);
  float*    red   = (float*)(smem + K2_RED);

  __builtin_prefetch(ft_W, 0, 3);
  __builtin_prefetch(gc_W, 0, 3);

  // async global->LDS edge lists (ASYNCcnt path)
  for (int e = tid; e < EPG_; e += 256) {
    async_g2l_b32((unsigned)(K2_PSRC + e * 4), patient_src + e);
    async_g2l_b32((unsigned)(K2_PDST + e * 4), patient_dst + e);
  }
  if (tid < 128) { maskf[tid] = (float)mask[tid]; dego[tid] = 0.f; degi[tid] = 0.f; }
  wait_async0();
  __syncthreads();
  for (int e = tid; e < EPG_; e += 256) {
    atomicAdd(&dego[psrc[e]], 1.f);
    atomicAdd(&degi[pdst[e]], 1.f);
  }
  __syncthreads();
  if (tid < 128) { dego[tid] = fmaxf(dego[tid], 1.f); degi[tid] = fmaxf(degi[tid], 1.f); }

  // ---- ft GEMM (WMMA): x[128,768] @ ft_W[768,128], K staged in 6 chunks ----
  v8f accs[8];
#pragma unroll
  for (int i = 0; i < 8; ++i) accs[i] = vzero8();
  for (int kc = 0; kc < 6; ++kc) {
    __syncthreads();
    for (int idx = tid; idx < 128 * 128; idx += 256) {
      int n = idx >> 7, kk = idx & 127;
      int k = kc * 128 + kk;
      float v = (k < ODIM_) ? orig[n * ODIM_ + k] : ws_rep[(size_t)n * 512 + (k - ODIM_)];
      bufA[idx] = (_Float16)v;
    }
    for (int idx = tid; idx < 128 * 128; idx += 256) {  // swizzled ft_W chunk
      int t = idx >> 9, ln = (idx >> 4) & 31, i = idx & 15;
      int kt = t >> 3, nt = t & 7;
      int k = kc * 128 + kt * 32 + ((ln >> 4) << 4) + i;
      int n = nt * 16 + (ln & 15);
      bufB[idx] = (_Float16)ft_W[(size_t)k * 128 + n];
    }
    __syncthreads();
#pragma unroll
    for (int i = 0; i < 8; ++i) {
      int t = wave * 8 + i;
      int mt = t >> 3, nt = t & 7;
#pragma unroll
      for (int kt = 0; kt < 4; ++kt) {
        v16h a  = frag_a(bufA, 128, mt * 16, kt * 32, lane);
        v16h bf = frag_b_sw(bufB, kt * 8 + nt, lane);
        accs[i] = wmma16(a, bf, accs[i]);
      }
    }
  }
  __syncthreads();
#pragma unroll
  for (int i = 0; i < 8; ++i) {
    int t = wave * 8 + i;
    store_c_f32(tmp, 128, (t >> 3) * 16, (t & 7) * 16, lane, accs[i]);
  }
  __syncthreads();

  // ---- BN(ft) + ReLU -> h, hsum ----
  if (tid < 128) {
    const int c = tid;
    const float bb = ft_b[c];
    float sum = 0.f, sq = 0.f;
    for (int n = 0; n < 128; ++n) { float x = tmp[n * 128 + c] + bb; sum += x; sq += x * x; }
    float mu  = sum * (1.f / 128.f);
    float var = sq * (1.f / 128.f) - mu * mu;
    float inv = rsqrtf(var + EPSF);
    float gg = ft_g[c], bw = ft_be[c];
    for (int n = 0; n < 128; ++n) {
      float x = tmp[n * 128 + c] + bb;
      float y = gg * (x - mu) * inv + bw;
      y = y > 0.f ? y : 0.f;
      h[n * 128 + c] = y;
      hsum[n * 128 + c] = y;
    }
  }
  __syncthreads();

  // ---- 3 masked GraphConv layers ----
  for (int it = 0; it < 3; ++it) {
    for (int idx = tid; idx < 128 * 128; idx += 256) tmp[idx] = 0.f;
    __syncthreads();
    // wave-per-edge, lanes stripe channels/banks
    for (int e = wave; e < EPG_; e += 8) {
      int s = psrc[e], d = pdst[e];
      float ew = maskf[s] * maskf[d];
      if (ew != 0.f) {
        float rs = ew * rsqrtf(dego[s]);
#pragma unroll
        for (int j = 0; j < 4; ++j) {
          int c = lane + 32 * j;
          atomicAdd(&tmp[d * 128 + c], h[s * 128 + c] * rs);
        }
      }
    }
    __syncthreads();
    for (int idx = tid; idx < 128 * 128; idx += 256) {
      int n = idx >> 7;
      bufA[idx] = (_Float16)(tmp[idx] * rsqrtf(degi[n]));
    }
    for (int idx = tid; idx < 128 * 128; idx += 256) {  // swizzled gc_W[it]
      int t = idx >> 9, ln = (idx >> 4) & 31, i = idx & 15;
      int kt = t >> 3, nt = t & 7;
      int k = kt * 32 + ((ln >> 4) << 4) + i;
      int n = nt * 16 + (ln & 15);
      bufB[idx] = (_Float16)gc_W[it * 16384 + k * 128 + n];
    }
    __syncthreads();
#pragma unroll
    for (int i = 0; i < 8; ++i) {
      int t = wave * 8 + i;
      int mt = t >> 3, nt = t & 7;
      v8f acc = vzero8();
#pragma unroll
      for (int kt = 0; kt < 4; ++kt) {
        v16h a  = frag_a(bufA, 128, mt * 16, kt * 32, lane);
        v16h bf = frag_b_sw(bufB, kt * 8 + nt, lane);
        acc = wmma16(a, bf, acc);
      }
      store_c_f32(tmp, 128, mt * 16, nt * 16, lane, acc);
    }
    __syncthreads();
    if (tid < 128) {
      const int c = tid;
      const float bb = gc_b[it * 128 + c];
      float sum = 0.f, sq = 0.f;
      for (int n = 0; n < 128; ++n) { float x = tmp[n * 128 + c] + bb; sum += x; sq += x * x; }
      float mu  = sum * (1.f / 128.f);
      float var = sq * (1.f / 128.f) - mu * mu;
      float inv = rsqrtf(var + EPSF);
      float gg = gbn_g[it * 128 + c], bw = gbn_be[it * 128 + c];
      for (int n = 0; n < 128; ++n) {
        float x = tmp[n * 128 + c] + bb;
        float y = gg * (x - mu) * inv + bw;
        y = y > 0.f ? y : 0.f;
        float nh = h[n * 128 + c] + y;
        h[n * 128 + c] = nh;
        hsum[n * 128 + c] += nh;
      }
    }
    __syncthreads();
  }

  // ---- classifier GEMM (WMMA): (hsum/4)[128,128] @ c1_W[128,64] ----
  for (int idx = tid; idx < 128 * 128; idx += 256) bufA[idx] = (_Float16)(hsum[idx] * 0.25f);
  for (int idx = tid; idx < 128 * 64; idx += 256) {  // swizzled c1_W (NT=4)
    int t = idx >> 9, ln = (idx >> 4) & 31, i = idx & 15;
    int kt = t >> 2, nt = t & 3;
    int k = kt * 32 + ((ln >> 4) << 4) + i;
    int n = nt * 16 + (ln & 15);
    bufB[idx] = (_Float16)c1_W[k * 64 + n];
  }
  __syncthreads();
#pragma unroll
  for (int i = 0; i < 4; ++i) {
    int t = wave * 4 + i;
    int mt = t >> 2, nt = t & 3;
    v8f acc = vzero8();
#pragma unroll
    for (int kt = 0; kt < 4; ++kt) {
      v16h a  = frag_a(bufA, 128, mt * 16, kt * 32, lane);
      v16h bf = frag_b_sw(bufB, kt * 4 + nt, lane);
      acc = wmma16(a, bf, acc);
    }
    store_c_f32(z, 64, mt * 16, nt * 16, lane, acc);
  }
  __syncthreads();

  // ---- LayerNorm + ReLU + final 64->2 ----
  if (tid < 128) {
    const int n = tid;
    float sum = 0.f, sq = 0.f;
    for (int j = 0; j < 64; ++j) { float x = z[n * 64 + j] + c1_b[j]; sum += x; sq += x * x; }
    float mu  = sum * (1.f / 64.f);
    float var = sq * (1.f / 64.f) - mu * mu;
    float inv = rsqrtf(var + EPSF);
    float l0 = c2_b[0], l1 = c2_b[1];
    for (int j = 0; j < 64; ++j) {
      float x = z[n * 64 + j] + c1_b[j];
      float y = ln_g[j] * (x - mu) * inv + ln_be[j];
      y = y > 0.f ? y : 0.f;
      l0 += y * c2_W[j * 2 + 0];
      l1 += y * c2_W[j * 2 + 1];
    }
    out[n * 2 + 0] = l0;
    out[n * 2 + 1] = l1;
  }

  // ---- avg recon loss over 2048 plane graphs ----
  {
    float rs = 0.f;
    for (int i = tid; i < B_ * P_; i += 256) rs += ws_rloss[i];
    red[tid] = rs;
  }
  __syncthreads();
  for (int off = 128; off > 0; off >>= 1) {
    if (tid < off) red[tid] += red[tid + off];
    __syncthreads();
  }
  if (tid == 0) out[256] = red[0] * (1.f / (float)(B_ * P_));
}

// ---------------- host ----------------
extern "C" void kernel_launch(void* const* d_in, const int* in_sizes, int n_in,
                              void* d_out, int out_size, void* d_ws, size_t ws_size,
                              hipStream_t stream) {
  (void)in_sizes; (void)n_in; (void)out_size; (void)ws_size;
  const float* plane_feat  = (const float*)d_in[0];
  const int*   plane_src   = (const int*)d_in[1];
  const int*   plane_dst   = (const int*)d_in[2];
  const int*   patient_src = (const int*)d_in[3];
  const int*   patient_dst = (const int*)d_in[4];
  const float* orig        = (const float*)d_in[5];
  const int*   mask        = (const int*)d_in[6];
  const float* W1   = (const float*)d_in[7];
  const float* al1  = (const float*)d_in[8];
  const float* ar1  = (const float*)d_in[9];
  const float* res1 = (const float*)d_in[10];
  const float* b1   = (const float*)d_in[11];
  const float* g1   = (const float*)d_in[12];
  const float* be1  = (const float*)d_in[13];
  const float* W2   = (const float*)d_in[14];
  const float* al2  = (const float*)d_in[15];
  const float* ar2  = (const float*)d_in[16];
  const float* res2 = (const float*)d_in[17];
  const float* b2   = (const float*)d_in[18];
  const float* g2   = (const float*)d_in[19];
  const float* be2  = (const float*)d_in[20];
  const float* dW1  = (const float*)d_in[21];
  const float* db1  = (const float*)d_in[22];
  const float* dg   = (const float*)d_in[23];
  const float* dbe  = (const float*)d_in[24];
  const float* dW2  = (const float*)d_in[25];
  const float* db2  = (const float*)d_in[26];
  const float* ft_W = (const float*)d_in[27];
  const float* ft_b = (const float*)d_in[28];
  const float* ft_g = (const float*)d_in[29];
  const float* ft_be= (const float*)d_in[30];
  const float* gc_W = (const float*)d_in[31];
  const float* gc_b = (const float*)d_in[32];
  const float* gbn_g= (const float*)d_in[33];
  const float* gbn_be=(const float*)d_in[34];
  const float* c1_W = (const float*)d_in[35];
  const float* c1_b = (const float*)d_in[36];
  const float* ln_g = (const float*)d_in[37];
  const float* ln_be= (const float*)d_in[38];
  const float* c2_W = (const float*)d_in[39];
  const float* c2_b = (const float*)d_in[40];

  float* ws_rep   = (float*)d_ws;                   // [B * P*32] floats
  float* ws_rloss = ws_rep + (size_t)B_ * P_ * 32;  // [2048]
  float* out      = (float*)d_out;                  // 257 floats

  (void)hipFuncSetAttribute((const void*)plane_encoder_kernel,
                            hipFuncAttributeMaxDynamicSharedMemorySize, K1_SMEM);
  (void)hipFuncSetAttribute((const void*)patient_head_kernel,
                            hipFuncAttributeMaxDynamicSharedMemorySize, K2_SMEM);

  plane_encoder_kernel<<<dim3(B_ * P_), dim3(256), K1_SMEM, stream>>>(
      plane_feat, plane_src, plane_dst, W1, al1, ar1, res1, b1, g1, be1, W2, al2,
      ar2, res2, b2, g2, be2, dW1, db1, dg, dbe, dW2, db2, ws_rep, ws_rloss);

  patient_head_kernel<<<dim3(1), dim3(256), K2_SMEM, stream>>>(
      orig, patient_src, patient_dst, mask, ft_W, ft_b, ft_g, ft_be, gc_W, gc_b,
      gbn_g, gbn_be, c1_W, c1_b, ln_g, ln_be, c2_W, c2_b, ws_rep, ws_rloss, out);
}
